// OIMLoss_tri_43001212567993
// MI455X (gfx1250) — matrix operand order repaired and checked by reading, not compiled, not run on hardware
//
#include <hip/hip_runtime.h>
#include <hip/hip_bf16.h>
#include <math.h>

// ---------------------------------------------------------------------------
// OIM + triplet loss for MI455X (gfx1250, wave32, WMMA).
// GEMMs run on v_wmma_f32_16x16x32_f16 with fp32 accumulation; banks are
// streamed from HBM once per pass (memory-bound at 23.3 TB/s), x stays in
// registers in WMMA A-operand layout. Double-buffered LDS staging pipelines
// the next tile's HBM fetch behind the current tile's WMMA chain; a rolling
// 2-deep B-operand register buffer overlaps ds_load latency with the WMMAs.
// ---------------------------------------------------------------------------

typedef _Float16 half_t;
typedef __attribute__((ext_vector_type(16))) _Float16 v16h;
typedef __attribute__((ext_vector_type(8)))  float    v8f;

#define N_SAMPLES 100000
#define N_FEAT    256
#define BATCH     256
#define NTILES    (N_SAMPLES / 16)   // 6250 column tiles of 16
#define TEMP_INV  20.0f              // 1 / 0.05
#define MARGIN_F  0.1f
#define EPS_F     0.1f
#define SHIFT_F   20.0f              // logit shift for exp (logits <= 20)
#define GEMM_GRID 1024
#define FENC_NEGINF 0x007FFFFFu      // fenc(-inf)

// order-preserving float<->uint encoding for atomicMax on floats
__device__ __forceinline__ unsigned fenc(float f) {
  unsigned u = __float_as_uint(f);
  return (u & 0x80000000u) ? ~u : (u | 0x80000000u);
}
__device__ __forceinline__ float fdec(unsigned u) {
  return (u & 0x80000000u) ? __uint_as_float(u & 0x7FFFFFFFu)
                           : __uint_as_float(~u);
}

// ---------------------------------------------------------------------------
// 0) per-call accumulator init (harness does not re-zero ws between replays)
// ---------------------------------------------------------------------------
__global__ void oim_init_kernel(float* S_exp, float* S_sum, unsigned* maxpos,
                                unsigned* maxneg, float* lossRow) {
  int r = threadIdx.x;
  S_exp[r]   = 0.0f;
  S_sum[r]   = 0.0f;
  lossRow[r] = 0.0f;
  maxpos[r]  = FENC_NEGINF;
  maxneg[r]  = FENC_NEGINF;
}

// ---------------------------------------------------------------------------
// 1) normalize x rows; emit fp32 copy + f16 copy pre-swizzled into the WMMA
//    A-operand layout:  block(mtile,ks) -> 32 lanes x 16 halves, where
//    lane<16 holds K {0..15} and lane>=16 holds K {16..31}.
// ---------------------------------------------------------------------------
__global__ void oim_prep_kernel(const float* __restrict__ inputs,
                                half_t* __restrict__ xf16,
                                float* __restrict__ xf32) {
  int r = blockIdx.x;    // batch row
  int k = threadIdx.x;   // feature
  float v = inputs[r * N_FEAT + k];
  __shared__ float red[256];
  red[k] = v * v;
  __syncthreads();
  for (int s = 128; s > 0; s >>= 1) {
    if (k < s) red[k] += red[k + s];
    __syncthreads();
  }
  float xn = v / sqrtf(red[0]);
  xf32[r * N_FEAT + k] = xn;

  int mtile = r >> 4, lrow = r & 15;
  int ks = k >> 5, ko = k & 31;
  int lane, e;
  if      (ko <  8) { lane = lrow;      e = ko;      }
  else if (ko < 16) { lane = lrow;      e = ko;      }
  else if (ko < 24) { lane = lrow + 16; e = ko - 16; }
  else              { lane = lrow + 16; e = ko - 16; }
  xf16[(((mtile * 8 + ks) * 32 + lane) << 4) + e] = (half_t)xn;
}

// ---------------------------------------------------------------------------
// GEMM worker. 512 threads = 16 waves; wave w owns M-tile w (rows 16w..16w+15)
// with its 8 A operands resident in registers. Grid-strided loop over 6250
// N-tiles, double-buffered: while tile t's 8 chained WMMAs run out of LDS
// buffer p, tile t+1's bank rows are in flight from HBM into registers and
// are converted/stored into buffer 1-p after the epilogue (one barrier/tile).
//   MODE 0 (CE):   r0 += exp(20*sim - 20), r1 += 20*sim
//   MODE 1 (MAX):  r0 = max over positives, r1 = max over negatives
//   MODE 2 (LOSS): r0 += pos_sel*(1-sim) + neg_sel*sim
// C layout: VGPR j, lanes 0-15 -> (row 16w+j, col n0+lane); lanes 16-31 ->
// (row 16w+8+j, col n0+lane-16).
// ---------------------------------------------------------------------------
template <int MODE>
__global__ void __launch_bounds__(512)
oim_gemm_kernel(const float* __restrict__ bank,
                const half_t* __restrict__ xf16,
                const long long* __restrict__ targets,
                const long long* __restrict__ labels,
                float* __restrict__ accA,        // MODE0: S_exp  MODE2: lossRow
                float* __restrict__ accB,        // MODE0: S_sum
                unsigned* __restrict__ maxpos,   // MODE1
                unsigned* __restrict__ maxneg,   // MODE1
                const float* __restrict__ thr_pos,  // MODE2
                const float* __restrict__ thr_neg)  // MODE2
{
  __shared__ __attribute__((aligned(64))) half_t ldsB[2][16 * N_FEAT]; // 16 KB
  __shared__ int ldsLab[2][16];

  const int tid  = threadIdx.x;
  const int lane = tid & 31;
  const int w    = tid >> 5;     // wave id == M-tile
  const int hs   = lane >> 4;    // lane-half select
  const int ln16 = lane & 15;
  const int sn   = tid >> 5;     // staging: bank row within tile
  const int skc  = (tid & 31) * 8;  // staging: 8 floats per thread

  // A operands (f16, pre-swizzled) -> registers, reused for all 6250 tiles
  v16h a[8];
  const v16h* xa = (const v16h*)xf16;
#pragma unroll
  for (int ks = 0; ks < 8; ++ks) a[ks] = xa[(w * 8 + ks) * 32 + lane];

  int   tgt[8];
  float thp[8], thn[8];
  if (MODE != 0) {
#pragma unroll
    for (int j = 0; j < 8; ++j) {
      int row = w * 16 + hs * 8 + j;
      tgt[j] = (int)targets[row];            // values < 1e5, fits int32
      if (MODE == 2) { thp[j] = thr_pos[row]; thn[j] = thr_neg[row]; }
    }
  }

  float r0[8], r1[8];
#pragma unroll
  for (int j = 0; j < 8; ++j) {
    r0[j] = (MODE == 1) ? -__builtin_inff() : 0.0f;
    r1[j] = (MODE == 1) ? -__builtin_inff() : 0.0f;
  }

  // ---- prologue: stage first tile into buffer 0 ----
  int tile = blockIdx.x;   // grid 1024 <= NTILES, so always valid
  {
    const float4* src =
        (const float4*)(bank + (size_t)(tile * 16 + sn) * N_FEAT + skc);
    float4 g0 = src[0];
    float4 g1 = src[1];
    half_t* dst = &ldsB[0][sn * N_FEAT + skc];
    dst[0] = (half_t)g0.x; dst[1] = (half_t)g0.y;
    dst[2] = (half_t)g0.z; dst[3] = (half_t)g0.w;
    dst[4] = (half_t)g1.x; dst[5] = (half_t)g1.y;
    dst[6] = (half_t)g1.z; dst[7] = (half_t)g1.w;
    if (MODE != 0 && tid < 16) ldsLab[0][tid] = (int)labels[tile * 16 + tid];
  }

  int buf = 0;
  for (; tile < NTILES; tile += gridDim.x) {
    __syncthreads();  // staged data for this tile is visible

    // ---- issue next tile's HBM loads; they fly during the WMMA chain ----
    const int nt = tile + gridDim.x;
    const bool havenext = nt < NTILES;
    float4 g0, g1;
    int nlab = 0;
    if (havenext) {
      const float4* src =
          (const float4*)(bank + (size_t)(nt * 16 + sn) * N_FEAT + skc);
      g0 = src[0];
      g1 = src[1];
      if (MODE != 0 && tid < 16) nlab = (int)labels[nt * 16 + tid];
    }

    // ---- compute: 8 chained WMMAs, rolling 2-deep B prefetch ----
    const half_t* bp = &ldsB[buf][ln16 * N_FEAT + hs * 16];
    v8f acc = {0.f, 0.f, 0.f, 0.f, 0.f, 0.f, 0.f, 0.f};
    v16h b0 = *(const v16h*)(bp);
    v16h b1;
#pragma unroll
    for (int ks = 0; ks < 8; ++ks) {
      if (ks + 1 < 8) {   // issue next B load before current WMMA
        v16h nb = *(const v16h*)(bp + (ks + 1) * 32);
        if (ks & 1) b0 = nb; else b1 = nb;
      }
      v16h cur = (ks & 1) ? b1 : b0;
      acc = __builtin_amdgcn_wmma_f32_16x16x32_f16(
          false, a[ks], false, cur, (short)0, acc, false, false);
    }

    // ---- epilogue on the 16x16 C tile ----
    const int lab = (MODE != 0) ? ldsLab[buf][ln16] : 0;
#pragma unroll
    for (int j = 0; j < 8; ++j) {
      float sim = acc[j];
      if (MODE == 0) {
        float lg = sim * TEMP_INV;
        r0[j] += __expf(lg - SHIFT_F);
        r1[j] += lg;
      } else if (MODE == 1) {
        bool pos = (lab == tgt[j]);
        r0[j] = pos ? fmaxf(r0[j], sim) : r0[j];
        r1[j] = pos ? r1[j] : fmaxf(r1[j], sim);
      } else {
        bool pos = (lab == tgt[j]);
        float c = pos ? ((sim < thp[j]) ? (1.0f - sim) : 0.0f)
                      : ((sim > thn[j]) ? sim : 0.0f);
        r0[j] += c;
      }
    }

    // ---- convert + store next tile into the other buffer (no barrier:
    //      writes go to buf^1 while everyone reads buf) ----
    if (havenext) {
      half_t* dst = &ldsB[buf ^ 1][sn * N_FEAT + skc];
      dst[0] = (half_t)g0.x; dst[1] = (half_t)g0.y;
      dst[2] = (half_t)g0.z; dst[3] = (half_t)g0.w;
      dst[4] = (half_t)g1.x; dst[5] = (half_t)g1.y;
      dst[6] = (half_t)g1.z; dst[7] = (half_t)g1.w;
      if (MODE != 0 && tid < 16) ldsLab[buf ^ 1][tid] = nlab;
    }
    buf ^= 1;
  }

  // reduce across the 16 lanes sharing each row, then commit via atomics
#pragma unroll
  for (int j = 0; j < 8; ++j) {
    float v0 = r0[j], v1 = r1[j];
#pragma unroll
    for (int m = 1; m < 16; m <<= 1) {
      float o0 = __shfl_xor(v0, m, 32);
      float o1 = __shfl_xor(v1, m, 32);
      if (MODE == 1) { v0 = fmaxf(v0, o0); v1 = fmaxf(v1, o1); }
      else           { v0 += o0;           v1 += o1;           }
    }
    if (ln16 == 0) {  // lane 0 -> row 16w+j ; lane 16 -> row 16w+8+j
      int row = w * 16 + hs * 8 + j;
      if (MODE == 0) {
        atomicAdd(&accA[row], v0);
        atomicAdd(&accB[row], v1);
      } else if (MODE == 1) {
        atomicMax(&maxpos[row], fenc(v0));
        atomicMax(&maxneg[row], fenc(v1));
      } else {
        atomicAdd(&accA[row], v0);
      }
    }
  }
}

// ---------------------------------------------------------------------------
// 3) logit at the target column: logit_t[b] = 20 * dot(x[b], features[t[b]])
// ---------------------------------------------------------------------------
__global__ void oim_tdot_kernel(const float* __restrict__ xf32,
                                const float* __restrict__ features,
                                const long long* __restrict__ targets,
                                float* __restrict__ logit_t) {
  int b = blockIdx.x, k = threadIdx.x;
  long long t = targets[b];
  __shared__ float red[256];
  red[k] = xf32[b * N_FEAT + k] * features[(size_t)t * N_FEAT + k];
  __syncthreads();
  for (int s = 128; s > 0; s >>= 1) {
    if (k < s) red[k] += red[k + s];
    __syncthreads();
  }
  if (k == 0) logit_t[b] = red[0] * TEMP_INV;
}

// ---------------------------------------------------------------------------
// 5) per-row triplet thresholds from the max pass
// ---------------------------------------------------------------------------
__global__ void oim_thresh_kernel(const unsigned* maxpos, const unsigned* maxneg,
                                  float* thr_pos, float* thr_neg) {
  int r = threadIdx.x;
  float mp = fdec(maxpos[r]);
  float mn = fdec(maxneg[r]);
  thr_pos[r] = mn + MARGIN_F;                 // pos_sel: sim < max_neg+margin
  thr_neg[r] = fmaxf(0.6f, mp) - MARGIN_F;    // neg_sel: sim > thr
}

// ---------------------------------------------------------------------------
// 7) final scalar losses
// ---------------------------------------------------------------------------
__global__ void oim_final_kernel(const float* S_exp, const float* S_sum,
                                 const float* logit_t, const unsigned* maxpos,
                                 const float* lossRow, float* out) {
  int r = threadIdx.x;
  float lse    = logf(S_exp[r]) + SHIFT_F;
  float lp_t   = logit_t[r] - lse;
  float sum_lp = S_sum[r] - (float)N_SAMPLES * lse;
  float ce = -((1.0f - EPS_F) * lp_t + (EPS_F / (float)N_SAMPLES) * sum_lp);
  float l2 = (maxpos[r] != FENC_NEGINF) ? lossRow[r] : 0.0f;  // has_pos gate
  __shared__ float ra[256], rb[256];
  ra[r] = ce; rb[r] = l2;
  __syncthreads();
  for (int s = 128; s > 0; s >>= 1) {
    if (r < s) { ra[r] += ra[r + s]; rb[r] += rb[r + s]; }
    __syncthreads();
  }
  if (r == 0) {
    out[0] = ra[0] / (float)BATCH;
    out[1] = rb[0] / (float)BATCH;
  }
}

// ---------------------------------------------------------------------------
// launch
// ---------------------------------------------------------------------------
extern "C" void kernel_launch(void* const* d_in, const int* in_sizes, int n_in,
                              void* d_out, int out_size, void* d_ws,
                              size_t ws_size, hipStream_t stream) {
  const float*     inputs   = (const float*)d_in[0];
  const float*     features = (const float*)d_in[1];
  const float*     sfeat    = (const float*)d_in[2];
  const long long* targets  = (const long long*)d_in[3];
  const long long* slabels  = (const long long*)d_in[4];

  char* ws = (char*)d_ws;                       // ~402 KB used
  half_t*   xf16    = (half_t*)(ws + 0);        // 128 KB, A-operand layout
  float*    xf32    = (float*)(ws + 131072);    // 256 KB
  float*    S_exp   = (float*)(ws + 393216);
  float*    S_sum   = (float*)(ws + 394240);
  float*    logit_t = (float*)(ws + 395264);
  unsigned* maxpos  = (unsigned*)(ws + 396288);
  unsigned* maxneg  = (unsigned*)(ws + 397312);
  float*    thr_pos = (float*)(ws + 398336);
  float*    thr_neg = (float*)(ws + 399360);
  float*    lossRow = (float*)(ws + 400384);
  float*    out     = (float*)d_out;

  oim_init_kernel<<<1, 256, 0, stream>>>(S_exp, S_sum, maxpos, maxneg, lossRow);
  oim_prep_kernel<<<BATCH, 256, 0, stream>>>(inputs, xf16, xf32);

  oim_gemm_kernel<0><<<GEMM_GRID, 512, 0, stream>>>(
      features, xf16, targets, slabels, S_exp, S_sum,
      maxpos, maxneg, thr_pos, thr_neg);

  oim_tdot_kernel<<<BATCH, 256, 0, stream>>>(xf32, features, targets, logit_t);

  oim_gemm_kernel<1><<<GEMM_GRID, 512, 0, stream>>>(
      sfeat, xf16, targets, slabels, S_exp, S_sum,
      maxpos, maxneg, thr_pos, thr_neg);

  oim_thresh_kernel<<<1, 256, 0, stream>>>(maxpos, maxneg, thr_pos, thr_neg);

  oim_gemm_kernel<2><<<GEMM_GRID, 512, 0, stream>>>(
      sfeat, xf16, targets, slabels, lossRow, S_sum,
      maxpos, maxneg, thr_pos, thr_neg);

  oim_final_kernel<<<1, 256, 0, stream>>>(S_exp, S_sum, logit_t, maxpos,
                                          lossRow, out);
}
